// GraphConvolutionalNet_78889959292957
// MI455X (gfx1250) — matrix operand order repaired and checked by reading, not compiled, not run on hardware
//
#include <hip/hip_runtime.h>
#include <hip/hip_bf16.h>
#include <math.h>

// ---------------------------------------------------------------------------
// GCN on gfx1250: bf16 WMMA GEMMs + fp32 atomic scatter aggregation.
// ---------------------------------------------------------------------------

typedef __bf16 bf16_t;
typedef __attribute__((ext_vector_type(16))) __bf16 v16bf;
typedef __attribute__((ext_vector_type(8)))  __bf16 v8bf;
typedef __attribute__((ext_vector_type(8)))  float  v8f;

#define N_NODES 50000
#define F_IN    256
#define F_HID   128
#define F_OUT   16
#define LDS_STRIDE 264   // 256 + 8 halves padding; row = 528 B (16B aligned)

// ---- weight convert + transpose: W[k][n] f32 -> Wt[n][k] bf16 -------------
__global__ __launch_bounds__(256) void convert_wt(const float* __restrict__ W,
                                                  bf16_t* __restrict__ Wt,
                                                  int K, int N) {
  int t = blockIdx.x * blockDim.x + threadIdx.x;
  if (t >= K * N) return;
  int n = t / K;
  int k = t - n * K;
  Wt[t] = (bf16_t)W[k * N + n];
}

// ---- degree: deg[col[e]] += ew[e] -----------------------------------------
__global__ __launch_bounds__(256) void deg_kernel(const long long* __restrict__ col,
                                                  const float* __restrict__ ew,
                                                  float* __restrict__ deg, int E) {
  int e = blockIdx.x * blockDim.x + threadIdx.x;
  if (e < E) atomicAdd(&deg[(int)col[e]], ew[e]);
}

__global__ __launch_bounds__(256) void dinv_kernel(const float* __restrict__ deg,
                                                   float* __restrict__ dinv, int N) {
  int i = blockIdx.x * blockDim.x + threadIdx.x;
  if (i < N) {
    float d = deg[i];
    dinv[i] = (d > 0.0f) ? rsqrtf(d) : 0.0f;
  }
}

// ---- GEMM1: h[50000x128] = x[50000x256] @ W1, bf16 WMMA, fp32 accum -------
// Block = 8 waves; block computes 16 rows x 128 cols. Wave w -> cols w*16..+15.
__global__ __launch_bounds__(256) void gemm_xW1(const float* __restrict__ x,
                                                const bf16_t* __restrict__ Wt1,  // [128][256]
                                                float* __restrict__ h) {
  __shared__ bf16_t As[16 * LDS_STRIDE];
  const int tid = threadIdx.x;
  const int block_row = blockIdx.x * 16;

  // stage x tile fp32 -> bf16 into LDS (coalesced: 256 consecutive floats/iter)
  #pragma unroll
  for (int i = 0; i < 16; ++i) {
    float v = x[(size_t)(block_row + i) * F_IN + tid];
    As[i * LDS_STRIDE + tid] = (bf16_t)v;
  }
  __syncthreads();

  const int wave = tid >> 5;
  const int lane = tid & 31;
  const int m    = lane & 15;   // A row within tile
  const int half = lane >> 4;   // lane half selects K grouping
  const int n    = lane & 15;   // B/C column
  const int nbase = wave * 16;

  v8f acc = {};
  for (int kb = 0; kb < F_IN; kb += 32) {
    // A fragment: lane holds K = {half*8..+7} U {16+half*8..+7} of row m
    v8bf a_lo = *(const v8bf*)&As[m * LDS_STRIDE + kb + half * 8];
    v8bf a_hi = *(const v8bf*)&As[m * LDS_STRIDE + kb + half * 8 + 16];
    v16bf A;
    #pragma unroll
    for (int i = 0; i < 8; ++i) { A[i] = a_lo[i]; A[i + 8] = a_hi[i]; }
    // B fragment: lane holds K = {half*8..+7} U {16+half*8..+7} of column nbase+n (Wt1 is [n][k])
    const bf16_t* wp = &Wt1[(size_t)(nbase + n) * F_IN + kb + half * 8];
    v8bf b_lo = *(const v8bf*)wp;
    v8bf b_hi = *(const v8bf*)(wp + 16);
    v16bf B;
    #pragma unroll
    for (int i = 0; i < 8; ++i) { B[i] = b_lo[i]; B[i + 8] = b_hi[i]; }

    acc = __builtin_amdgcn_wmma_f32_16x16x32_bf16(false, A, false, B,
                                                  (short)0, acc, false, false);
  }
  // C layout: VGPR r, lanes 0-15 -> M=r, lanes 16-31 -> M=r+8; N = lane&15
  #pragma unroll
  for (int r = 0; r < 8; ++r)
    h[(size_t)(block_row + r + half * 8) * F_HID + nbase + n] = acc[r];
}

// ---- edge aggregation F=128: out[col] += norm * h[row] (wave per edge) ----
__global__ __launch_bounds__(256) void aggregate_f128(const long long* __restrict__ row,
                                                      const long long* __restrict__ col,
                                                      const float* __restrict__ ew,
                                                      const float* __restrict__ dinv,
                                                      const float* __restrict__ h,
                                                      float* __restrict__ out, int E) {
  int gw     = (blockIdx.x * blockDim.x + threadIdx.x) >> 5;
  int lane   = threadIdx.x & 31;
  int nwaves = (gridDim.x * blockDim.x) >> 5;
  for (int e = gw; e < E; e += nwaves) {
    int r = (int)row[e];
    int c = (int)col[e];
    float w = dinv[r] * ew[e] * dinv[c];
    const float4 v = *(const float4*)&h[(size_t)r * F_HID + lane * 4];
    float* dst = &out[(size_t)c * F_HID + lane * 4];
    atomicAdd(dst + 0, w * v.x);
    atomicAdd(dst + 1, w * v.y);
    atomicAdd(dst + 2, w * v.z);
    atomicAdd(dst + 3, w * v.w);
  }
}

// ---- h1 = relu(agg1 + b1) -> bf16 (row-major, WMMA-A-ready) ---------------
__global__ __launch_bounds__(256) void relu_bias_bf16(const float* __restrict__ agg,
                                                      const float* __restrict__ b,
                                                      bf16_t* __restrict__ h1b, int total) {
  int i = blockIdx.x * blockDim.x + threadIdx.x;
  if (i >= total) return;
  float v = agg[i] + b[i & (F_HID - 1)];
  h1b[i] = (bf16_t)fmaxf(v, 0.0f);
}

// ---- GEMM2: h2[50000x16] = h1b[50000x128] @ W2, wave per 16-row tile ------
__global__ __launch_bounds__(256) void gemm_hW2(const bf16_t* __restrict__ h1b,
                                                const bf16_t* __restrict__ Wt2,  // [16][128]
                                                float* __restrict__ h2, int row_tiles) {
  int wave = threadIdx.x >> 5;
  int lane = threadIdx.x & 31;
  int rt   = blockIdx.x * 8 + wave;
  if (rt >= row_tiles) return;   // wave-uniform exit: EXEC stays all-ones for WMMA
  const int m    = lane & 15;
  const int half = lane >> 4;
  const int n    = lane & 15;

  const bf16_t* arow = h1b + (size_t)(rt * 16 + m) * F_HID;
  v8f acc = {};
  #pragma unroll
  for (int kb = 0; kb < F_HID; kb += 32) {
    v8bf a_lo = *(const v8bf*)(arow + kb + half * 8);
    v8bf a_hi = *(const v8bf*)(arow + kb + half * 8 + 16);
    v16bf A;
    #pragma unroll
    for (int i = 0; i < 8; ++i) { A[i] = a_lo[i]; A[i + 8] = a_hi[i]; }
    const bf16_t* wp = Wt2 + (size_t)n * F_HID + kb + half * 8;
    v8bf b_lo = *(const v8bf*)wp;
    v8bf b_hi = *(const v8bf*)(wp + 16);
    v16bf B;
    #pragma unroll
    for (int i = 0; i < 8; ++i) { B[i] = b_lo[i]; B[i + 8] = b_hi[i]; }
    acc = __builtin_amdgcn_wmma_f32_16x16x32_bf16(false, A, false, B,
                                                  (short)0, acc, false, false);
  }
  #pragma unroll
  for (int r = 0; r < 8; ++r)
    h2[(size_t)(rt * 16 + r + half * 8) * F_OUT + n] = acc[r];
}

// ---- edge aggregation F=16 (wave per edge, 16 active lanes) ---------------
__global__ __launch_bounds__(256) void aggregate_f16(const long long* __restrict__ row,
                                                     const long long* __restrict__ col,
                                                     const float* __restrict__ ew,
                                                     const float* __restrict__ dinv,
                                                     const float* __restrict__ h2,
                                                     float* __restrict__ out, int E) {
  int gw     = (blockIdx.x * blockDim.x + threadIdx.x) >> 5;
  int lane   = threadIdx.x & 31;
  int nwaves = (gridDim.x * blockDim.x) >> 5;
  for (int e = gw; e < E; e += nwaves) {
    int r = (int)row[e];
    int c = (int)col[e];
    float w = dinv[r] * ew[e] * dinv[c];
    if (lane < F_OUT)
      atomicAdd(&out[(size_t)c * F_OUT + lane], w * h2[(size_t)r * F_OUT + lane]);
  }
}

// ---- out = log_softmax(agg2 + b2, axis=1), 16 classes ---------------------
__global__ __launch_bounds__(256) void logsoftmax16(const float* __restrict__ agg,
                                                    const float* __restrict__ b,
                                                    float* __restrict__ out, int N) {
  int i = blockIdx.x * blockDim.x + threadIdx.x;
  if (i >= N) return;
  float v[F_OUT];
  float mx = -INFINITY;
  #pragma unroll
  for (int j = 0; j < F_OUT; ++j) {
    v[j] = agg[(size_t)i * F_OUT + j] + b[j];
    mx = fmaxf(mx, v[j]);
  }
  float s = 0.0f;
  #pragma unroll
  for (int j = 0; j < F_OUT; ++j) s += expf(v[j] - mx);
  float lse = mx + logf(s);
  #pragma unroll
  for (int j = 0; j < F_OUT; ++j) out[(size_t)i * F_OUT + j] = v[j] - lse;
}

// ---------------------------------------------------------------------------
static inline size_t align_up(size_t x) { return (x + 255) & ~(size_t)255; }

extern "C" void kernel_launch(void* const* d_in, const int* in_sizes, int n_in,
                              void* d_out, int out_size, void* d_ws, size_t ws_size,
                              hipStream_t stream) {
  const float*     x   = (const float*)d_in[0];
  const long long* ei  = (const long long*)d_in[1];   // int64 [2, E]
  const float*     ew  = (const float*)d_in[2];
  const float*     W1  = (const float*)d_in[3];
  const float*     b1  = (const float*)d_in[4];
  const float*     W2  = (const float*)d_in[5];
  const float*     b2  = (const float*)d_in[6];
  float*           out = (float*)d_out;

  const int E = in_sizes[2];
  const long long* row = ei;
  const long long* col = ei + E;

  // workspace carve-up (all offsets 256B-aligned)
  char* ws = (char*)d_ws;
  size_t off = 0;
  bf16_t* Wt1  = (bf16_t*)(ws + off); off = align_up(off + (size_t)F_IN  * F_HID * 2);
  bf16_t* Wt2  = (bf16_t*)(ws + off); off = align_up(off + (size_t)F_HID * F_OUT * 2);
  float*  h    = (float*) (ws + off); off = align_up(off + (size_t)N_NODES * F_HID * 4);
  float*  deg  = (float*) (ws + off); off = align_up(off + (size_t)N_NODES * 4);
  float*  dinv = (float*) (ws + off); off = align_up(off + (size_t)N_NODES * 4);
  float*  agg1 = (float*) (ws + off); off = align_up(off + (size_t)N_NODES * F_HID * 4);
  bf16_t* h1b  = (bf16_t*)(ws + off); off = align_up(off + (size_t)N_NODES * F_HID * 2);
  float*  h2   = (float*) (ws + off); off = align_up(off + (size_t)N_NODES * F_OUT * 4);
  float*  agg2 = (float*) (ws + off); off = align_up(off + (size_t)N_NODES * F_OUT * 4);
  (void)ws_size;

  // zero accumulators (graph-capture-safe stream ops)
  hipMemsetAsync(deg,  0, (size_t)N_NODES * 4, stream);
  hipMemsetAsync(agg1, 0, (size_t)N_NODES * F_HID * 4, stream);
  hipMemsetAsync(agg2, 0, (size_t)N_NODES * F_OUT * 4, stream);

  // weights -> bf16 transposed [n][k]
  convert_wt<<<(F_IN * F_HID + 255) / 256, 256, 0, stream>>>(W1, Wt1, F_IN, F_HID);
  convert_wt<<<(F_HID * F_OUT + 255) / 256, 256, 0, stream>>>(W2, Wt2, F_HID, F_OUT);

  // normalization
  deg_kernel<<<(E + 255) / 256, 256, 0, stream>>>(col, ew, deg, E);
  dinv_kernel<<<(N_NODES + 255) / 256, 256, 0, stream>>>(deg, dinv, N_NODES);

  // layer 1
  gemm_xW1<<<N_NODES / 16, 256, 0, stream>>>(x, Wt1, h);
  aggregate_f128<<<(E + 7) / 8, 256, 0, stream>>>(row, col, ew, dinv, h, agg1, E);
  relu_bias_bf16<<<(N_NODES * F_HID + 255) / 256, 256, 0, stream>>>(agg1, b1, h1b,
                                                                    N_NODES * F_HID);
  // layer 2
  const int row_tiles = N_NODES / 16;  // 3125
  gemm_hW2<<<(row_tiles + 7) / 8, 256, 0, stream>>>(h1b, Wt2, h2, row_tiles);
  aggregate_f16<<<(E + 7) / 8, 256, 0, stream>>>(row, col, ew, dinv, h2, agg2, E);

  // output
  logsoftmax16<<<(N_NODES + 255) / 256, 256, 0, stream>>>(agg2, b2, out, N_NODES);
  (void)out_size; (void)n_in;
}
